// Model_87960930222087
// MI455X (gfx1250) — matrix operand order repaired and checked by reading, not compiled, not run on hardware
//
#include <hip/hip_runtime.h>
#include <math.h>

// Model dims
#define DMODEL 256
#define SEQL   1024
#define NB     16
#define NH     8
#define DHD    32
#define DFFN   1024
#define NU     35
#define MROWS  (NB * SEQL)   // 16384

typedef __attribute__((ext_vector_type(16))) __bf16        v16bf;
typedef __attribute__((ext_vector_type(8)))  float         v8f;
typedef __attribute__((ext_vector_type(4)))  float         f4;
typedef __attribute__((ext_vector_type(8)))  unsigned short v8us;
typedef __attribute__((ext_vector_type(16))) unsigned short v16us;
typedef __attribute__((ext_vector_type(4)))  unsigned int   v4u;

__device__ __forceinline__ unsigned short f32_to_bf16(float f) {
  unsigned int u = __builtin_bit_cast(unsigned int, f);
  u += 0x7FFFu + ((u >> 16) & 1u);          // round-to-nearest-even
  return (unsigned short)(u >> 16);
}
__device__ __forceinline__ float gelu_exact(float x) {
  return 0.5f * x * (1.0f + erff(x * 0.7071067811865475f));
}
__device__ __forceinline__ unsigned hash_u32(unsigned x) {
  x ^= x >> 16; x *= 0x7feb352du; x ^= x >> 15; x *= 0x846ca68bu; x ^= x >> 16;
  return x;
}
__device__ __forceinline__ v8f wmma_bf16(const v16us& a, const v16us& b, v8f c) {
  return __builtin_amdgcn_wmma_f32_16x16x32_bf16(
      false, __builtin_bit_cast(v16bf, a), false, __builtin_bit_cast(v16bf, b),
      (short)0, c, false, false);
}

// ---------------------------------------------------------------------------
// Weight convert+transpose: W[K,N] f32 -> WT[N,K] bf16 (once per launch)
// ---------------------------------------------------------------------------
__global__ __launch_bounds__(256) void k_w2bf(const float* __restrict__ W,
                                              unsigned short* __restrict__ WT,
                                              int K, int N) {
  int i = blockIdx.x * 256 + threadIdx.x;   // coalesced read over K*N
  int k = i / N, n = i % N;
  WT[(size_t)n * K + k] = f32_to_bf16(W[i]);
}

// ---------------------------------------------------------------------------
// Tiled WMMA GEMM:  C[M,N] = A[M,K] @ B[K,N]  (+bias, +gelu, +residual)
// B is pre-transposed bf16: Bt[N,K].
// MODE 0: A row-major f32 from global (vectorized b128 staging).
// MODE 1: embedding conv — A row (b,s), col k=(w*256+c) gathers
//         x_enc[b, c*SEQ + (s+w-1 mod L)]; epilogue adds sinusoidal PE.
// Block: 256 threads (8 waves). Block tile 128x64, K-step 32, LDS
// double-buffered (one barrier per K-step). Wave w owns a 32x32 sub-tile:
// four v_wmma_f32_16x16x32_bf16 per K-step.
// ---------------------------------------------------------------------------
template <int MODE>
__global__ __launch_bounds__(256) void k_gemm(
    const float* __restrict__ A, const unsigned short* __restrict__ Bt,
    const float* __restrict__ bias, const float* __restrict__ Res,
    float* __restrict__ C, int M, int N, int K, int opt) {
  __shared__ unsigned short As[2][128][40];   // +8 pad keeps 16B row alignment
  __shared__ unsigned short Bs[2][64][40];    // transposed: Bs[.][n][k]

  const int tid  = threadIdx.x;
  const int lane = tid & 31;
  const int wave = tid >> 5;
  const int rowBase = blockIdx.y * 128;
  const int colBase = blockIdx.x * 64;
  const int subM = (wave >> 1) * 32;
  const int subN = (wave & 1) * 32;

  v8f acc00 = {}, acc01 = {}, acc10 = {}, acc11 = {};

  // stage K-tile k0 into LDS buffer bf (A: 128x32 via b128, B: 64x32 via b128)
  const int c0 = (tid & 3) * 8;      // octet within the 32-wide K tile
  const int rb = tid >> 2;           // 0..63
  auto stage = [&](int k0, int bf) {
#pragma unroll
    for (int i = 0; i < 2; ++i) {
      int r = rb + i * 64;
      float e[8];
      if (MODE == 0) {
        const f4* ap = (const f4*)&A[(size_t)(rowBase + r) * K + k0 + c0];
        f4 f0 = ap[0], f1 = ap[1];
#pragma unroll
        for (int j = 0; j < 4; ++j) { e[j] = f0[j]; e[j + 4] = f1[j]; }
      } else {
        int gr = rowBase + r;
        int b  = gr >> 10, s = gr & 1023;
#pragma unroll
        for (int j = 0; j < 8; ++j) {
          int kk = k0 + c0 + j;
          int w  = kk >> 8, ch = kk & 255;
          int sp = (s + w - 1 + SEQL) & (SEQL - 1);
          e[j] = A[((size_t)b << 18) + ((size_t)ch << 10) + sp];
        }
      }
      v4u packed;
#pragma unroll
      for (int j = 0; j < 4; ++j)
        packed[j] = (unsigned)f32_to_bf16(e[2 * j]) |
                    ((unsigned)f32_to_bf16(e[2 * j + 1]) << 16);
      *(v4u*)&As[bf][r][c0] = packed;
    }
    // B: one b128 load + one b128 store per thread (bf16, pre-transposed)
    *(v4u*)&Bs[bf][rb][c0] =
        *(const v4u*)&Bt[(size_t)(colBase + rb) * K + k0 + c0];
  };

  const int iters = K >> 5;
  stage(0, 0);
  __syncthreads();

  for (int it = 0; it < iters; ++it) {
    const int p = it & 1;
    // stage next K-tile into the other buffer (writes go to buf p^1 while
    // everyone reads buf p; __syncthreads() waits DScnt before the barrier)
    if (it + 1 < iters) stage(((it + 1) << 5), p ^ 1);
    // prefetch the K+2 tile into cache (global_prefetch_b8)
    if (MODE == 0 && it + 2 < iters) {
      int k2 = (it + 2) << 5;
      if (tid < 128) __builtin_prefetch(&A[(size_t)(rowBase + tid) * K + k2], 0, 3);
      else if (tid < 192) __builtin_prefetch(&Bt[(size_t)(colBase + tid - 128) * K + k2], 0, 3);
    }

    // ---- A fragments (16x32 bf16): lanes 0-15 row M, K {0..7,16..23};
    //      lanes 16-31 same rows, K {8..15,24..31} ----
    const int am  = subM + (lane & 15);
    const int akb = (lane >> 4) << 3;                 // 0 or 8
    v16us av0, av1;
    {
      v8us lo = *(v8us*)&As[p][am][akb];
      v8us hi = *(v8us*)&As[p][am][akb + 16];
#pragma unroll
      for (int i = 0; i < 8; ++i) { av0[i] = lo[i]; av0[i + 8] = hi[i]; }
    }
    {
      v8us lo = *(v8us*)&As[p][am + 16][akb];
      v8us hi = *(v8us*)&As[p][am + 16][akb + 16];
#pragma unroll
      for (int i = 0; i < 8; ++i) { av1[i] = lo[i]; av1[i + 8] = hi[i]; }
    }
    // ---- B fragments (32x16 bf16): lane = col, K {0..7,16..23} / {8..15,24..31} by half ----
    const int bn  = subN + (lane & 15);
    const int bko = (lane >> 4) << 3;                 // 0 or 8
    v16us bv0, bv1;
    {
      v8us lo = *(v8us*)&Bs[p][bn][bko];
      v8us hi = *(v8us*)&Bs[p][bn][bko + 16];
#pragma unroll
      for (int i = 0; i < 8; ++i) { bv0[i] = lo[i]; bv0[i + 8] = hi[i]; }
    }
    {
      v8us lo = *(v8us*)&Bs[p][bn + 16][bko];
      v8us hi = *(v8us*)&Bs[p][bn + 16][bko + 16];
#pragma unroll
      for (int i = 0; i < 8; ++i) { bv1[i] = lo[i]; bv1[i + 8] = hi[i]; }
    }

    acc00 = wmma_bf16(av0, bv0, acc00);
    acc01 = wmma_bf16(av0, bv1, acc01);
    acc10 = wmma_bf16(av1, bv0, acc10);
    acc11 = wmma_bf16(av1, bv1, acc11);
    __syncthreads();
  }

  // ---- epilogue: C/D layout — VGPR r: lanes0-15 M=r, lanes16-31 M=8+r ----
#pragma unroll
  for (int mi = 0; mi < 2; ++mi) {
    int m0 = rowBase + subM + mi * 16 + ((lane >> 4) << 3);
#pragma unroll
    for (int ni = 0; ni < 2; ++ni) {
      int col = colBase + subN + ni * 16 + (lane & 15);
      v8f accv = mi ? (ni ? acc11 : acc10) : (ni ? acc01 : acc00);
#pragma unroll
      for (int r = 0; r < 8; ++r) {
        int row = m0 + r;
        float vv = accv[r];
        if (MODE == 0) {
          if (bias) vv += bias[col];
          if (opt & 1) vv = gelu_exact(vv);
          if (opt & 2) vv += Res[(size_t)row * N + col];
        } else {
          int s = row & (SEQL - 1);
          float ang = (float)s * __expf(-(float)((col >> 1) << 1) *
                                        (9.210340371976184f / (float)DMODEL));
          float sv = __sinf(ang), cv = __cosf(ang);
          vv += (col & 1) ? cv : sv;    // branchless select (v_cndmask)
        }
        C[(size_t)row * N + col] = vv;
      }
    }
  }
}

// ---------------------------------------------------------------------------
// LayerNorm over last dim (256); one block per row.
// ---------------------------------------------------------------------------
__global__ __launch_bounds__(256) void k_ln(const float* __restrict__ X,
                                            const float* __restrict__ g,
                                            const float* __restrict__ bt,
                                            float* __restrict__ Y) {
  __shared__ float red[256];
  const int row = blockIdx.x, tid = threadIdx.x;
  float v = X[(size_t)row * DMODEL + tid];
  red[tid] = v; __syncthreads();
  for (int s = 128; s > 0; s >>= 1) { if (tid < s) red[tid] += red[tid + s]; __syncthreads(); }
  float mean = red[0] * (1.0f / DMODEL);
  __syncthreads();
  float dv = v - mean;
  red[tid] = dv * dv; __syncthreads();
  for (int s = 128; s > 0; s >>= 1) { if (tid < s) red[tid] += red[tid + s]; __syncthreads(); }
  float var = red[0] * (1.0f / DMODEL);
  Y[(size_t)row * DMODEL + tid] = dv * rsqrtf(var + 1e-5f) * g[tid] + bt[tid];
}

// ---------------------------------------------------------------------------
// ProbSparse sampled scores: M[b,h,l] = max_u(q.k_idx) - mean_u(q.k_idx)
// ---------------------------------------------------------------------------
__global__ __launch_bounds__(256) void k_mscore(const float* __restrict__ Q,
                                                const float* __restrict__ Km,
                                                float* __restrict__ Mout) {
  int i = blockIdx.x * 256 + threadIdx.x;     // (b*H + h)*L + l
  int l = i & (SEQL - 1);
  int bh = i >> 10, h = bh & (NH - 1), b = bh >> 3;
  const float* qp = Q + ((size_t)b << 18) + (size_t)l * DMODEL + h * DHD;
  float qr[DHD];
#pragma unroll
  for (int d = 0; d < DHD; ++d) qr[d] = qp[d];
  float mx = -1e30f, sm = 0.0f;
  for (int u = 0; u < NU; ++u) {
    unsigned kk = hash_u32((unsigned)(l * 64 + u) + 0x9e3779b9u) & (SEQL - 1);
    const float* kp = Km + ((size_t)b << 18) + (size_t)kk * DMODEL + h * DHD;
    float s = 0.0f;
#pragma unroll
    for (int d = 0; d < DHD; ++d) s += qr[d] * kp[d];
    mx = fmaxf(mx, s); sm += s;
  }
  Mout[i] = mx - sm * (1.0f / NU);
}

// ---------------------------------------------------------------------------
// Iterative top-35 argmax per (b,h); one block per (b,h).
// ---------------------------------------------------------------------------
__global__ __launch_bounds__(256) void k_topk(const float* __restrict__ Min,
                                              int* __restrict__ Mtop) {
  __shared__ float vals[SEQL];
  __shared__ float rmx[256];
  __shared__ int   rix[256];
  const int bh = blockIdx.x, tid = threadIdx.x;
#pragma unroll
  for (int j = 0; j < 4; ++j) vals[tid + j * 256] = Min[(size_t)bh * SEQL + tid + j * 256];
  __syncthreads();
  for (int it = 0; it < NU; ++it) {
    float best = -1e30f; int bidx = 0;
#pragma unroll
    for (int j = 0; j < 4; ++j) {
      int e = tid + j * 256;
      float v = vals[e];
      if (v > best) { best = v; bidx = e; }
    }
    rmx[tid] = best; rix[tid] = bidx; __syncthreads();
    for (int s = 128; s > 0; s >>= 1) {
      if (tid < s && rmx[tid + s] > rmx[tid]) { rmx[tid] = rmx[tid + s]; rix[tid] = rix[tid + s]; }
      __syncthreads();
    }
    if (tid == 0) { Mtop[bh * NU + it] = rix[0]; vals[rix[0]] = -1e30f; }
    __syncthreads();
  }
}

// mean of V over sequence per (b,h,d)
__global__ __launch_bounds__(256) void k_meanv(const float* __restrict__ V,
                                               float* __restrict__ meanv) {
  int i = blockIdx.x * 256 + threadIdx.x;     // b*H*DH
  int d = i & 31, h = (i >> 5) & 7, b = i >> 8;
  const float* vp = V + ((size_t)b << 18) + h * DHD + d;
  float s = 0.0f;
  for (int l = 0; l < SEQL; ++l) s += vp[(size_t)l * DMODEL];
  meanv[i] = s * (1.0f / SEQL);
}

// broadcast mean(V) into ctx (B,L,D)
__global__ __launch_bounds__(256) void k_ctxfill(const float* __restrict__ meanv,
                                                 float* __restrict__ ctx) {
  size_t i = (size_t)blockIdx.x * 256 + threadIdx.x;
  int hd = (int)(i & 255);
  int b  = (int)(i >> 18);
  ctx[i] = meanv[(b * NH + (hd >> 5)) * DHD + (hd & 31)];
}

// full softmax attention for the NU selected queries; one block per (b,h,u)
__global__ __launch_bounds__(256) void k_attnupd(const float* __restrict__ Q,
                                                 const float* __restrict__ Km,
                                                 const float* __restrict__ V,
                                                 const int* __restrict__ Mtop,
                                                 float* __restrict__ Ctx) {
  __shared__ float qr[DHD];
  __shared__ float sc[SEQL];
  __shared__ float red[256];
  __shared__ float pv[8][DHD];
  const int tid = threadIdx.x;
  const int bhu = blockIdx.x;
  const int u = bhu % NU, bh = bhu / NU, h = bh & 7, b = bh >> 3;
  const int lq = Mtop[bh * NU + u];
  if (tid < DHD) qr[tid] = Q[((size_t)b << 18) + (size_t)lq * DMODEL + h * DHD + tid];
  __syncthreads();
#pragma unroll
  for (int j = 0; j < 4; ++j) {
    int l = tid + j * 256;
    const float* kp = Km + ((size_t)b << 18) + (size_t)l * DMODEL + h * DHD;
    float s = 0.0f;
#pragma unroll
    for (int d = 0; d < DHD; ++d) s += qr[d] * kp[d];
    sc[l] = s * 0.17677669529663687f;   // 1/sqrt(32)
  }
  __syncthreads();
  float mx = -1e30f;
#pragma unroll
  for (int j = 0; j < 4; ++j) mx = fmaxf(mx, sc[tid + j * 256]);
  red[tid] = mx; __syncthreads();
  for (int s = 128; s > 0; s >>= 1) { if (tid < s) red[tid] = fmaxf(red[tid], red[tid + s]); __syncthreads(); }
  mx = red[0]; __syncthreads();
  float ls = 0.0f;
#pragma unroll
  for (int j = 0; j < 4; ++j) {
    int l = tid + j * 256;
    float p = __expf(sc[l] - mx);
    sc[l] = p; ls += p;
  }
  red[tid] = ls; __syncthreads();
  for (int s = 128; s > 0; s >>= 1) { if (tid < s) red[tid] += red[tid + s]; __syncthreads(); }
  const float S = red[0];
  const int d = tid & 31, chunk = tid >> 5;
  float acc = 0.0f;
  for (int l = chunk * 128; l < chunk * 128 + 128; ++l)
    acc += sc[l] * V[((size_t)b << 18) + (size_t)l * DMODEL + h * DHD + d];
  pv[chunk][d] = acc; __syncthreads();
  if (tid < DHD) {
    float t = 0.0f;
#pragma unroll
    for (int c = 0; c < 8; ++c) t += pv[c][tid];
    Ctx[((size_t)b << 18) + (size_t)lq * DMODEL + h * DHD + tid] = t / S;
  }
}

// final: partial sums of gelu(h) * xm * Wp per (batch, segment)
__global__ __launch_bounds__(256) void k_finpart(const float* __restrict__ Hb,
                                                 const float* __restrict__ XM,
                                                 const float* __restrict__ Wp,
                                                 float* __restrict__ part) {
  __shared__ float red[256];
  const int tid = threadIdx.x, b = blockIdx.y, seg = blockIdx.x;
  float acc = 0.0f;
#pragma unroll
  for (int j = 0; j < 16; ++j) {
    int e = seg * 4096 + j * 256 + tid;            // e = s*256 + d
    float hv = gelu_exact(Hb[((size_t)b << 18) + e]);
    int s = e >> 8, d = e & 255;
    float xmv = XM[((size_t)b << 18) + ((size_t)d << 10) + s];
    acc += hv * xmv * Wp[e];
  }
  red[tid] = acc; __syncthreads();
  for (int s = 128; s > 0; s >>= 1) { if (tid < s) red[tid] += red[tid + s]; __syncthreads(); }
  if (tid == 0) part[b * 64 + seg] = red[0];
}

__global__ __launch_bounds__(64) void k_finred(const float* __restrict__ part,
                                               const float* __restrict__ bp,
                                               float* __restrict__ out) {
  int t = threadIdx.x;
  if (t < NB) {
    float s = bp[0];
    for (int j = 0; j < 64; ++j) s += part[t * 64 + j];
    out[t] = s;
  }
}

// ---------------------------------------------------------------------------
extern "C" void kernel_launch(void* const* d_in, const int* in_sizes, int n_in,
                              void* d_out, int out_size, void* d_ws, size_t ws_size,
                              hipStream_t stream) {
  // pytree (sorted-key) flatten order of setup_inputs():
  // 0 x_enc, 1 x_mark_enc, 2 x_dec, 3 x_mark_dec,
  // 4 Wp, 5 bF, 6 bp, 7 emb_W, 8 gF, 9.. layers (16 leaves each, sorted):
  //   W1,W2,Wk,Wo,Wq,Wv,b1,b2,bc1,bc2,bk,bo,bq,bv,g1,g2
  const float* x_enc  = (const float*)d_in[0];
  const float* x_mark = (const float*)d_in[1];
  const float* Wp     = (const float*)d_in[4];
  const float* bF     = (const float*)d_in[5];
  const float* bp     = (const float*)d_in[6];
  const float* emb_W  = (const float*)d_in[7];
  const float* gF     = (const float*)d_in[8];

  float* ws  = (float*)d_ws;
  const size_t AN = (size_t)MROWS * DMODEL;        // 4,194,304
  float* h    = ws;
  float* q    = h   + AN;
  float* k    = q   + AN;
  float* v    = k   + AN;
  float* ctx  = v   + AN;
  float* t    = ctx + AN;
  float* ff   = t   + AN;                          // 16,777,216
  float* Mbuf = ff  + (size_t)MROWS * DFFN;        // 131,072
  int*   Mtop = (int*)(Mbuf + (size_t)NB * NH * SEQL);
  float* mnv  = (float*)(Mtop + NB * NH * NU);     // 4,096
  float* part = mnv + NB * NH * DHD;               // 1,024
  // bf16 transposed weights (us elements)
  unsigned short* embT = (unsigned short*)(part + 64 * NB);
  unsigned short* lT[2][6];                        // Wq,Wk,Wv,Wo,W1,W2 per layer
  {
    unsigned short* pw = embT + 768 * 256;
    for (int layer = 0; layer < 2; ++layer) {
      for (int widx = 0; widx < 4; ++widx) { lT[layer][widx] = pw; pw += 256 * 256; }
      lT[layer][4] = pw; pw += 256 * 1024;         // W1T (1024 x 256)
      lT[layer][5] = pw; pw += 1024 * 256;         // W2T (256 x 1024)
    }
  }
  float* outf = (float*)d_out;

  const dim3 blk(256);
  const dim3 g256(DMODEL / 64, MROWS / 128);       // (4, 128)
  const dim3 g1024(DFFN / 64, MROWS / 128);        // (16, 128)

  // ---- weight convert + transpose (f32 -> bf16, N x K) ----
  k_w2bf<<<768 * 256 / 256, blk, 0, stream>>>(emb_W, embT, 768, 256);
  for (int layer = 0; layer < 2; ++layer) {
    const int base = 9 + layer * 16;
    k_w2bf<<<256 * 256 / 256, blk, 0, stream>>>((const float*)d_in[base + 4], lT[layer][0], 256, 256);   // Wq
    k_w2bf<<<256 * 256 / 256, blk, 0, stream>>>((const float*)d_in[base + 2], lT[layer][1], 256, 256);   // Wk
    k_w2bf<<<256 * 256 / 256, blk, 0, stream>>>((const float*)d_in[base + 5], lT[layer][2], 256, 256);   // Wv
    k_w2bf<<<256 * 256 / 256, blk, 0, stream>>>((const float*)d_in[base + 3], lT[layer][3], 256, 256);   // Wo
    k_w2bf<<<256 * 1024 / 256, blk, 0, stream>>>((const float*)d_in[base + 0], lT[layer][4], 256, 1024); // W1
    k_w2bf<<<1024 * 256 / 256, blk, 0, stream>>>((const float*)d_in[base + 1], lT[layer][5], 1024, 256); // W2
  }

  // embedding conv (wrap) + positional encoding
  k_gemm<1><<<g256, blk, 0, stream>>>(x_enc, embT, nullptr, nullptr, h,
                                      MROWS, DMODEL, 3 * DMODEL, 0);

  for (int layer = 0; layer < 2; ++layer) {
    const int base = 9 + layer * 16;
    const float* b1  = (const float*)d_in[base + 6];
    const float* b2  = (const float*)d_in[base + 7];
    const float* bc1 = (const float*)d_in[base + 8];
    const float* bc2 = (const float*)d_in[base + 9];
    const float* bk  = (const float*)d_in[base + 10];
    const float* bo  = (const float*)d_in[base + 11];
    const float* bq  = (const float*)d_in[base + 12];
    const float* bv  = (const float*)d_in[base + 13];
    const float* g1  = (const float*)d_in[base + 14];
    const float* g2  = (const float*)d_in[base + 15];

    // projections (WMMA)
    k_gemm<0><<<g256, blk, 0, stream>>>(h, lT[layer][0], bq, nullptr, q, MROWS, DMODEL, DMODEL, 0);
    k_gemm<0><<<g256, blk, 0, stream>>>(h, lT[layer][1], bk, nullptr, k, MROWS, DMODEL, DMODEL, 0);
    k_gemm<0><<<g256, blk, 0, stream>>>(h, lT[layer][2], bv, nullptr, v, MROWS, DMODEL, DMODEL, 0);

    // ProbSparse attention
    k_mscore <<<MROWS * NH / 256, blk, 0, stream>>>(q, k, Mbuf);
    k_topk   <<<NB * NH, blk, 0, stream>>>(Mbuf, Mtop);
    k_meanv  <<<NB * NH * DHD / 256, blk, 0, stream>>>(v, mnv);
    k_ctxfill<<<(unsigned)(AN / 256), blk, 0, stream>>>(mnv, ctx);
    k_attnupd<<<NB * NH * NU, blk, 0, stream>>>(q, k, v, Mtop, ctx);

    // out proj + residual, LN1
    k_gemm<0><<<g256, blk, 0, stream>>>(ctx, lT[layer][3], bo, h, t, MROWS, DMODEL, DMODEL, 2);
    k_ln<<<MROWS, blk, 0, stream>>>(t, g1, b1, q);     // q := ln1 output

    // FFN (WMMA) with exact GELU, residual, LN2
    k_gemm<0><<<g1024, blk, 0, stream>>>(q, lT[layer][4], bc1, nullptr, ff, MROWS, DFFN, DMODEL, 1);
    k_gemm<0><<<g256,  blk, 0, stream>>>(ff, lT[layer][5], bc2, q, t, MROWS, DMODEL, DFFN, 2);
    k_ln<<<MROWS, blk, 0, stream>>>(t, g2, b2, h);
  }

  // final LN, gelu * mask, dot with Wp
  k_ln<<<MROWS, blk, 0, stream>>>(h, gF, bF, t);
  k_finpart<<<dim3(64, NB), blk, 0, stream>>>(t, x_mark, Wp, part);
  k_finred<<<1, 64, 0, stream>>>(part, bp, outf);
}
